// QCNN_Diff_12756052869705
// MI455X (gfx1250) — compile-verified, matrix operand order checked by reading, not run on hardware
//
#include <hip/hip_runtime.h>

typedef __attribute__((ext_vector_type(16))) _Float16 v16h;
typedef __attribute__((ext_vector_type(8)))  float    v8f;

#define NQ  8
#define DIM 256
#define HALF_PER_MAT (DIM * DIM)

// ---------------------------------------------------------------------------
// Kernel 1: build per-branch 256x256 unitaries (Re/Im, f16, row-major [i][j])
// One wave per basis column j; state held 8 complex amps per lane:
//   flat index idx = (lane<<3) | local ; wire w <-> bit (7-w) of idx
// ---------------------------------------------------------------------------

// gate g[8] = {u00r,u00i, u01r,u01i, u10r,u10i, u11r,u11i}
template <int TW, int CW>
__device__ __forceinline__ void apply_gate(float (&sre)[8], float (&sim)[8],
                                           const float g[8], int lane) {
  constexpr int BT = 7 - TW;                 // target bit position
  if constexpr (BT >= 3) {                   // target is a lane bit
    constexpr int LB = BT - 3;
    const int mybit = (lane >> LB) & 1;
#pragma unroll
    for (int l = 0; l < 8; ++l) {
      float pr = __shfl_xor(sre[l], 1 << LB, 32);
      float pi = __shfl_xor(sim[l], 1 << LB, 32);
      float ar = sre[l], ai = sim[l];
      float nr, ni;
      if (mybit == 0) {                      // new a0 = u00*a0 + u01*a1
        nr = g[0]*ar - g[1]*ai + g[2]*pr - g[3]*pi;
        ni = g[0]*ai + g[1]*ar + g[2]*pi + g[3]*pr;
      } else {                               // new a1 = u10*a0 + u11*a1
        nr = g[4]*pr - g[5]*pi + g[6]*ar - g[7]*ai;
        ni = g[4]*pi + g[5]*pr + g[6]*ai + g[7]*ar;
      }
      bool act = true;
      if constexpr (CW >= 0) {
        constexpr int BC = 7 - CW;
        int cb;
        if constexpr (BC >= 3) cb = (lane >> (BC - 3)) & 1;
        else                   cb = (l >> BC) & 1;
        act = (cb != 0);
      }
      if (act) { sre[l] = nr; sim[l] = ni; }
    }
  } else {                                   // target is a local bit
    constexpr int MASK = 1 << BT;
#pragma unroll
    for (int l0 = 0; l0 < 8; ++l0) {
      if ((l0 & MASK) != 0) continue;
      constexpr int DUMMY = 0; (void)DUMMY;
      const int l1 = l0 | MASK;
      bool act = true;
      if constexpr (CW >= 0) {
        constexpr int BC = 7 - CW;
        int cb;
        if constexpr (BC >= 3) cb = (lane >> (BC - 3)) & 1;
        else                   cb = (l0 >> BC) & 1;
        act = (cb != 0);
      }
      float a0r = sre[l0], a0i = sim[l0];
      float a1r = sre[l1], a1i = sim[l1];
      float n0r = g[0]*a0r - g[1]*a0i + g[2]*a1r - g[3]*a1i;
      float n0i = g[0]*a0i + g[1]*a0r + g[2]*a1i + g[3]*a1r;
      float n1r = g[4]*a0r - g[5]*a0i + g[6]*a1r - g[7]*a1i;
      float n1i = g[4]*a0i + g[5]*a0r + g[6]*a1i + g[7]*a1r;
      if (act) { sre[l0]=n0r; sim[l0]=n0i; sre[l1]=n1r; sim[l1]=n1i; }
    }
  }
}

__device__ __forceinline__ void make_cr(int kind, float t, float g[8]) {
  float c = __cosf(t * 0.5f), s = __sinf(t * 0.5f);
#pragma unroll
  for (int i = 0; i < 8; ++i) g[i] = 0.f;
  if (kind == 0)      { g[0]=c;           g[3]=-s; g[5]=-s; g[6]=c; }          // rx
  else if (kind == 1) { g[0]=c; g[2]=-s;  g[4]= s; g[6]=c; }                   // ry
  else                { g[0]=c; g[1]=-s;  g[6]=c;  g[7]=s; }                   // rz
}

__device__ __forceinline__ void make_u3(const float* p, float g[8]) {
  float ct = __cosf(p[0]*0.5f), st = __sinf(p[0]*0.5f);
  float cp = __cosf(p[1]), sp = __sinf(p[1]);
  float cl = __cosf(p[2]), sl = __sinf(p[2]);
  g[0] = ct;        g[1] = 0.f;
  g[2] = -cl * st;  g[3] = -sl * st;
  g[4] =  cp * st;  g[5] =  sp * st;
  float cpl = cp*cl - sp*sl, spl = sp*cl + cp*sl;   // e^{i(ph+lm)}
  g[6] = cpl * ct;  g[7] = spl * ct;
}

__global__ __launch_bounds__(256) void build_unitaries(
    const float* __restrict__ crx, const float* __restrict__ u3x,
    const float* __restrict__ cry, const float* __restrict__ u3y,
    const float* __restrict__ crz, const float* __restrict__ u3z,
    _Float16* __restrict__ U) {
  const int wid  = (blockIdx.x * blockDim.x + threadIdx.x) >> 5;  // 0..767
  const int lane = threadIdx.x & 31;
  const int br   = wid >> 8;     // 0=rx, 1=ry, 2=rz
  const int col  = wid & 255;    // basis column j

  const float* crp = (br == 0) ? crx : (br == 1) ? cry : crz;
  const float* u3p = (br == 0) ? u3x : (br == 1) ? u3y : u3z;

  float sre[8], sim[8];
#pragma unroll
  for (int l = 0; l < 8; ++l) { sre[l] = 0.f; sim[l] = 0.f; }
  if ((col >> 3) == lane) sre[col & 7] = 1.f;

  float g[8];
  // CR_PAIRS_1 = (0,1)(2,3)(4,5)(6,7)(1,2)(3,4)(5,6)
  make_cr(br, crp[0], g); apply_gate<1, 0>(sre, sim, g, lane);
  make_cr(br, crp[1], g); apply_gate<3, 2>(sre, sim, g, lane);
  make_cr(br, crp[2], g); apply_gate<5, 4>(sre, sim, g, lane);
  make_cr(br, crp[3], g); apply_gate<7, 6>(sre, sim, g, lane);
  make_cr(br, crp[4], g); apply_gate<2, 1>(sre, sim, g, lane);
  make_cr(br, crp[5], g); apply_gate<4, 3>(sre, sim, g, lane);
  make_cr(br, crp[6], g); apply_gate<6, 5>(sre, sim, g, lane);
  // U3 on wires 1,3,5,7
  make_u3(u3p + 0,  g); apply_gate<1, -1>(sre, sim, g, lane);
  make_u3(u3p + 3,  g); apply_gate<3, -1>(sre, sim, g, lane);
  make_u3(u3p + 6,  g); apply_gate<5, -1>(sre, sim, g, lane);
  make_u3(u3p + 9,  g); apply_gate<7, -1>(sre, sim, g, lane);
  // CR_PAIRS_2 = (1,3)(5,7)(3,5)
  make_cr(br, crp[7], g); apply_gate<3, 1>(sre, sim, g, lane);
  make_cr(br, crp[8], g); apply_gate<7, 5>(sre, sim, g, lane);
  make_cr(br, crp[9], g); apply_gate<5, 3>(sre, sim, g, lane);
  // U3 on wires 3, 7
  make_u3(u3p + 12, g); apply_gate<3, -1>(sre, sim, g, lane);
  make_u3(u3p + 15, g); apply_gate<7, -1>(sre, sim, g, lane);

  _Float16* Ure = U + (size_t)(br * 2 + 0) * HALF_PER_MAT;
  _Float16* Uim = U + (size_t)(br * 2 + 1) * HALF_PER_MAT;
#pragma unroll
  for (int l = 0; l < 8; ++l) {
    const int row = (lane << 3) | l;                 // output index i
    Ure[row * DIM + col] = (_Float16)sre[l];
    Uim[row * DIM + col] = (_Float16)sim[l];
  }
}

// ---------------------------------------------------------------------------
// Kernel 2: batched y = S * U^T via WMMA, |y|^2 -> signed feature sums -> MLP
// One wave handles 16 batch rows; 4 waves per block.
// ---------------------------------------------------------------------------

__device__ __forceinline__ float phi5(int h, const float* c, const float* s) {
  // product over bits 0..4 of h : bit p <-> wire (4-p)
  float p = ((h      ) & 1) ? s[4] : c[4];
  p *= ((h >> 1) & 1) ? s[3] : c[3];
  p *= ((h >> 2) & 1) ? s[2] : c[2];
  p *= ((h >> 3) & 1) ? s[1] : c[1];
  p *= ((h >> 4) & 1) ? s[0] : c[0];
  return p;
}

__global__ __launch_bounds__(128) void qcnn_main(
    const float* __restrict__ x, const _Float16* __restrict__ U,
    const float* __restrict__ w1, const float* __restrict__ b1,
    const float* __restrict__ w2, const float* __restrict__ b2,
    float* __restrict__ out) {
  __shared__ float sfeat[4][16][6];

  const int wave  = threadIdx.x >> 5;
  const int lane  = threadIdx.x & 31;
  const int tile  = blockIdx.x * 4 + wave;
  const int M     = lane & 15;          // A: row of tile; B: column N
  const int halfw = lane >> 4;          // half-wave select (K offset 8)
  const int kbase = halfw * 8;
  const int row   = tile * 16 + M;      // batch element for A operand

  // encoding factors for this batch row
  float c[8], s[8];
  const float* xr = x + row * NQ;
#pragma unroll
  for (int w = 0; w < 8; ++w) {
    float a = xr[w] * 0.5f;
    c[w] = __cosf(a); s[w] = __sinf(a);
  }
  // low products over flat bits 2..0 (wires 5,6,7)
  float Plo[8];
#pragma unroll
  for (int m = 0; m < 8; ++m) {
    float p = ((m >> 2) & 1) ? s[5] : c[5];
    p *= ((m >> 1) & 1) ? s[6] : c[6];
    p *= ( m       & 1) ? s[7] : c[7];
    Plo[m] = p;
  }
  // A fragments for the 8 K-tiles (16x32 f16 layout: lane=M, element e ->
  // K = kbase + e + (e>=8 ? 8 : 0);  j = kt*32 + K)
  v16h aall[8];
#pragma unroll
  for (int kt = 0; kt < 8; ++kt) {
    const float ph0 = phi5(kt * 4 + halfw,     c, s);   // elements 0..7
    const float ph1 = phi5(kt * 4 + 2 + halfw, c, s);   // elements 8..15
    v16h a;
#pragma unroll
    for (int e = 0; e < 8; ++e) a[e]     = (_Float16)(ph0 * Plo[e]);
#pragma unroll
    for (int e = 0; e < 8; ++e) a[e + 8] = (_Float16)(ph1 * Plo[e]);
    aall[kt] = a;
  }

  const float sign7 = (M & 1) ? -1.f : 1.f;   // wire7 <-> bit0 of column i
  v8f feat[6];
#pragma unroll
  for (int f = 0; f < 6; ++f) feat[f] = (v8f){};

#pragma unroll
  for (int br = 0; br < 3; ++br) {
    const _Float16* Ure = U + (size_t)(br * 2 + 0) * HALF_PER_MAT;
    const _Float16* Uim = U + (size_t)(br * 2 + 1) * HALF_PER_MAT;
    for (int nt = 0; nt < 16; ++nt) {
      const int i = nt * 16 + M;               // output column for this lane
      const _Float16* rre = Ure + i * DIM;
      const _Float16* rim = Uim + i * DIM;
      v8f cre = (v8f){}, cim = (v8f){};
#pragma unroll
      for (int kt = 0; kt < 8; ++kt) {
        const int j0 = kt * 32 + kbase;        // K-contiguous runs of 8
        union { v16h h; uint4 u[2]; } bre, bim;
        bre.u[0] = *(const uint4*)(rre + j0);
        bre.u[1] = *(const uint4*)(rre + j0 + 16);
        bim.u[0] = *(const uint4*)(rim + j0);
        bim.u[1] = *(const uint4*)(rim + j0 + 16);
        cre = __builtin_amdgcn_wmma_f32_16x16x32_f16(
            false, aall[kt], false, bre.h, (short)0, cre, false, false);
        cim = __builtin_amdgcn_wmma_f32_16x16x32_f16(
            false, aall[kt], false, bim.h, (short)0, cim, false, false);
      }
      const float sign3 = (nt & 1) ? -1.f : 1.f;  // wire3 <-> bit4 of i = nt&1
#pragma unroll
      for (int e = 0; e < 8; ++e) {
        const float p = cre[e] * cre[e] + cim[e] * cim[e];
        feat[br * 2 + 0][e] += sign3 * p;
        feat[br * 2 + 1][e] += sign7 * p;
      }
    }
  }

  // reduce over the 16 N-lanes within each half-wave (rows stay separate)
#pragma unroll
  for (int m = 1; m < 16; m <<= 1) {
#pragma unroll
    for (int f = 0; f < 6; ++f)
#pragma unroll
      for (int e = 0; e < 8; ++e)
        feat[f][e] += __shfl_xor(feat[f][e], m, 32);
  }

  // lane 0 holds rows 0..7 (element e), lane 16 holds rows 8..15
  if (M == 0) {
#pragma unroll
    for (int e = 0; e < 8; ++e)
#pragma unroll
      for (int f = 0; f < 6; ++f)
        sfeat[wave][halfw * 8 + e][f] = feat[f][e];
  }
  __syncthreads();

  // tiny MLP head: 16 lanes, one batch row each
  if (lane < 16) {
    float f[6];
#pragma unroll
    for (int k = 0; k < 6; ++k) f[k] = sfeat[wave][lane][k];
    float o = b2[0];
#pragma unroll
    for (int j = 0; j < 12; ++j) {
      float a = b1[j];
#pragma unroll
      for (int k = 0; k < 6; ++k) a += f[k] * w1[j * 6 + k];
      o += tanhf(a) * w2[j];
    }
    out[tile * 16 + lane] = 1.f / (1.f + __expf(-o));
  }
}

// ---------------------------------------------------------------------------
extern "C" void kernel_launch(void* const* d_in, const int* in_sizes, int n_in,
                              void* d_out, int out_size, void* d_ws, size_t ws_size,
                              hipStream_t stream) {
  const float* x   = (const float*)d_in[0];
  const float* crx = (const float*)d_in[1];
  const float* u3x = (const float*)d_in[2];
  const float* cry = (const float*)d_in[3];
  const float* u3y = (const float*)d_in[4];
  const float* crz = (const float*)d_in[5];
  const float* u3z = (const float*)d_in[6];
  const float* w1  = (const float*)d_in[7];
  const float* b1  = (const float*)d_in[8];
  const float* w2  = (const float*)d_in[9];
  const float* b2  = (const float*)d_in[10];
  float* out = (float*)d_out;
  _Float16* U = (_Float16*)d_ws;   // 3 branches * {re,im} * 256*256 f16 = 768 KB

  // Kernel 1: 3*256 = 768 waves -> 96 blocks of 256 threads
  build_unitaries<<<96, 256, 0, stream>>>(crx, u3x, cry, u3y, crz, u3z, U);

  const int n      = in_sizes[0] / NQ;   // batch size (32768)
  const int ntiles = n / 16;             // 2048 row tiles, 4 waves per block
  qcnn_main<<<ntiles / 4, 128, 0, stream>>>(x, U, w1, b1, w2, b2, out);
}
